// HoloSpectralBlock_532575944917
// MI455X (gfx1250) — compile-verified, matrix operand order checked by reading, not run on hardware
//
#include <hip/hip_runtime.h>
#include <hip/hip_bf16.h>

// HoloSpectralBlock on MI455X (gfx1250).
// FFT stages rewritten as circulant GEMMs on v_wmma_f32_16x16x32_f16.
// Circulant matrices materialized once as 8 MiB f16 (L2-resident, pre-
// transposed); GEMM staging is pure GLOBAL_LOAD_ASYNC_TO_LDS_B128 with a
// double-buffered pipeline using partial s_wait_asynccnt.
// Wave tile 64x64 (16 WMMA per 16 LDS fragment loads).

#define D_DIM 2048
#define S_DIM 2048
#define B_DIM 8
#define RANK  64
#define MASK  2047
#define LDA   40   // padded LDS row (halfs) for A tile (128x32)
#define LDB   40   // padded LDS row (halfs) for B^T tile (256x32)

#define USE_ASYNC_LDS 1

typedef _Float16 v16h __attribute__((ext_vector_type(16)));
typedef _Float16 v8h  __attribute__((ext_vector_type(8)));
typedef float    v8f  __attribute__((ext_vector_type(8)));

__device__ __forceinline__ float sigm(float x){ return 1.0f/(1.0f+__expf(-x)); }
__device__ __forceinline__ float gelu_tanh(float x){
  return 0.5f*x*(1.0f + tanhf(0.7978845608028654f*(x + 0.044715f*x*x*x)));
}

// 16-byte global->LDS copy (async on CDNA5; ASYNCcnt tracked)
__device__ __forceinline__ void async_copy16(const _Float16* g, _Float16* l){
#if USE_ASYNC_LDS
  unsigned loff = (unsigned)(size_t)l;                 // LDS byte offset
  unsigned long long ga = (unsigned long long)(size_t)g;
  asm volatile("global_load_async_to_lds_b128 %0, %1, off"
               :: "v"(loff), "v"(ga) : "memory");
#else
  *(v8h*)l = *(const v8h*)g;
#endif
}
// wait until at most NREM async ops remain (completion is in-order)
template<int NREM>
__device__ __forceinline__ void async_wait_le(){
#if USE_ASYNC_LDS
  if constexpr (NREM==0)      asm volatile("s_wait_asynccnt 0x0" ::: "memory");
  else if constexpr (NREM==2) asm volatile("s_wait_asynccnt 0x2" ::: "memory");
  else                        asm volatile("s_wait_asynccnt 0x6" ::: "memory");
#endif
}

// --------------------------------------------------- LayerNorm (f16 output)
__global__ void ln_kernel(const float* __restrict__ in, const float* __restrict__ w,
                          const float* __restrict__ b, _Float16* __restrict__ out,
                          float* __restrict__ mu_out, float* __restrict__ rs_out){
  const int tid = threadIdx.x;
  const size_t base = (size_t)blockIdx.x * D_DIM;
  float v[8]; float s=0.f, q=0.f;
  #pragma unroll
  for (int i=0;i<8;++i){ v[i] = in[base + tid + i*256]; s += v[i]; q += v[i]*v[i]; }
  __shared__ float r1[256], r2[256];
  r1[tid]=s; r2[tid]=q; __syncthreads();
  for (int off=128; off>0; off>>=1){
    if (tid<off){ r1[tid]+=r1[tid+off]; r2[tid]+=r2[tid+off]; }
    __syncthreads();
  }
  const float mean = r1[0]*(1.0f/D_DIM);
  const float var  = r2[0]*(1.0f/D_DIM) - mean*mean;
  const float rs   = rsqrtf(var + 1e-5f);
  #pragma unroll
  for (int i=0;i<8;++i){
    int d = tid + i*256;
    out[base + d] = (_Float16)((v[i]-mean)*rs*w[d] + b[d]);
  }
  if (mu_out != nullptr && tid==0){ mu_out[blockIdx.x]=mean; rs_out[blockIdx.x]=rs; }
}

// ------------------------------------------- kvec = irfft(sigmoid(freq_gate))
__global__ void kvec_kernel(const float* __restrict__ fg, float* __restrict__ kv){
  int n = blockIdx.x*256 + threadIdx.x;
  float s = sigm(fg[0]) + ((n&1)? -sigm(fg[1024]) : sigm(fg[1024]));
  for (int m=1; m<1024; ++m){
    float ph = (float)((m*n)&MASK) * (6.283185307179586f/2048.0f);
    s += 2.0f*sigm(fg[m])*__cosf(ph);
  }
  kv[n] = s*(1.0f/2048.0f);
}

// ----------------- materialize circulant matrix (pre-transposed), f16, 8 MiB
// dir=+1: out[r*2048+c] = key[(r-c) mod 2048]   (conv form)
// dir=-1: out[r*2048+c] = key[(c-r) mod 2048]   (corr form)
__global__ void build_circ(const float* __restrict__ key, _Float16* __restrict__ out,
                           int dir){
  const int r  = blockIdx.x;
  const int c8 = threadIdx.x * 8;
  v8h v;
  #pragma unroll
  for (int j=0;j<8;++j){
    int c = c8 + j;
    int idx = (dir > 0) ? ((r - c) & MASK) : ((c - r) & MASK);
    v[j] = (_Float16)key[idx];
  }
  *(v8h*)(out + (size_t)r*2048 + c8) = v;
}

// --------------------------------- cumsum along S on f16 buffer (fp32 accum)
__global__ void scan_phase_a(const _Float16* __restrict__ p, float* __restrict__ part){
  int bid = blockIdx.x;
  int dblk = bid & 7, ch = (bid>>3)&7, b = bid>>6;
  int d = dblk*256 + threadIdx.x;
  size_t colbase = ((size_t)b*S_DIM)*D_DIM + (size_t)(ch*256)*D_DIM + d;
  float s = 0.f;
  for (int i=0;i<256;++i) s += (float)p[colbase + (size_t)i*D_DIM];
  part[(b*8+ch)*D_DIM + d] = s;
}
__global__ void scan_phase_c(_Float16* __restrict__ p, const float* __restrict__ part){
  int bid = blockIdx.x;
  int dblk = bid & 7, ch = (bid>>3)&7, b = bid>>6;
  int d = dblk*256 + threadIdx.x;
  float acc = 0.f;
  for (int c=0;c<ch;++c) acc += part[(b*8+c)*D_DIM + d];
  size_t colbase = ((size_t)b*S_DIM)*D_DIM + (size_t)(ch*256)*D_DIM + d;
  for (int i=0;i<256;++i){
    size_t idx = colbase + (size_t)i*D_DIM;
    acc += (float)p[idx];
    p[idx] = (_Float16)acc;
  }
}

// ---------------------------------------------------------- WMMA GEMM core
// C(MxN) = A(MxK) * B(KxN), f16 operands, f32 accumulate.
// Block tile 128x256, 8 waves (2x4), wave tile 64x64 (4x4 WMMA subtiles).
// A: M x K row-major f16, async-staged.
// TRANSB=false: Bh is N x K row-major (pre-transposed), async-staged.
// TRANSB=true : Bh is K x N row-major (dense g), transpose-staged via LDS.
// EPI: 0 = store f16 to outh
//      1 = outp = x + h(recomputed LN1) + gate*acc   (f32)
//      2 = outp = res + acc                          (f32)
template<bool TRANSB, int EPI>
__global__ __launch_bounds__(256) void gemm_wmma(
    const _Float16* __restrict__ Ah, const _Float16* __restrict__ Bh,
    float* __restrict__ outp, _Float16* __restrict__ outh,
    int M, int N, int K,
    size_t zstrideB, size_t zstrideO,
    const float* __restrict__ xg, const float* __restrict__ mug,
    const float* __restrict__ rsg, const float* __restrict__ w1g,
    const float* __restrict__ b1g, const float* __restrict__ gateg,
    const float* __restrict__ resg)
{
  __shared__ __align__(16) _Float16 Abuf[2][128*LDA];
  __shared__ __align__(16) _Float16 Bbuf[2][256*LDB];

  const int tid  = threadIdx.x;
  const int lane = tid & 31;
  const int wid  = tid >> 5;
  const int wm   = wid >> 2;          // 0..1  (wave row)
  const int wn   = wid & 3;           // 0..3  (wave col)
  const int lm   = lane & 15;
  const int hk   = (lane >> 4) * 8;   // half-index base for K packing
  const int m0   = blockIdx.y * 128;
  const int n0   = blockIdx.x * 256;
  const size_t zB = (size_t)blockIdx.z * zstrideB;
  const size_t zO = (size_t)blockIdx.z * zstrideO;

  auto stage = [&](int k0, int buf){
    #pragma unroll
    for (int it=0; it<2; ++it){
      int idx = tid + it*256;
      int row = idx >> 2;            // 4 x 16B chunks per 32-half row
      int c8  = (idx & 3) * 8;
      async_copy16(Ah + (size_t)(m0+row)*K + k0 + c8, &Abuf[buf][row*LDA + c8]);
    }
    if constexpr (!TRANSB){
      #pragma unroll
      for (int it=0; it<4; ++it){
        int idx = tid + it*256;
        int row = idx >> 2;          // 256 n-rows
        int c8  = (idx & 3) * 8;
        async_copy16(Bh + (size_t)(n0+row)*K + k0 + c8, &Bbuf[buf][row*LDB + c8]);
      }
    } else {
      #pragma unroll
      for (int it=0; it<4; ++it){
        int idx = tid + it*256;
        int k   = idx >> 5;          // 32 x 8-half chunks per 256-half n-row
        int n8  = (idx & 31) * 8;
        const v8h v = *(const v8h*)(Bh + zB + (size_t)(k0+k)*N + n0 + n8);
        #pragma unroll
        for (int j=0;j<8;++j) Bbuf[buf][(n8+j)*LDB + k] = v[j];
      }
    }
  };

  v8f acc[4][4];
  #pragma unroll
  for (int mt=0; mt<4; ++mt)
    #pragma unroll
    for (int nt=0; nt<4; ++nt)
      acc[mt][nt] = (v8f){0.f,0.f,0.f,0.f,0.f,0.f,0.f,0.f};

  const int nsteps = K/32;
  stage(0, 0);
  for (int t=0; t<nsteps; ++t){
    const int cur = t & 1;
    if (t+1 < nsteps){
      stage((t+1)*32, cur^1);                 // prefetch next tile (async)
      async_wait_le<TRANSB ? 2 : 6>();        // current tile landed, next in flight
    } else {
      async_wait_le<0>();
    }
    __syncthreads();

    // fragments + WMMA (16-bit A layout: lanes0-15 K0..7/16..23, lanes16-31 +8)
    v16h af[4];
    #pragma unroll
    for (int mt=0; mt<4; ++mt){
      int row = wm*64 + mt*16 + lm;
      const v8h lo = *(const v8h*)&Abuf[cur][row*LDA + hk];
      const v8h hi = *(const v8h*)&Abuf[cur][row*LDA + hk + 16];
      #pragma unroll
      for (int i=0;i<8;++i){ af[mt][i]=lo[i]; af[mt][i+8]=hi[i]; }
    }
    #pragma unroll
    for (int nt=0; nt<4; ++nt){
      int col = wn*64 + nt*16 + lm;
      const v8h lo = *(const v8h*)&Bbuf[cur][col*LDB + hk];
      const v8h hi = *(const v8h*)&Bbuf[cur][col*LDB + hk + 16];
      v16h bf;
      #pragma unroll
      for (int i=0;i<8;++i){ bf[i]=lo[i]; bf[i+8]=hi[i]; }
      #pragma unroll
      for (int mt=0; mt<4; ++mt)
        acc[mt][nt] = __builtin_amdgcn_wmma_f32_16x16x32_f16(
            false, af[mt], false, bf, (short)0, acc[mt][nt], false, false);
    }
    __syncthreads();
  }

  // ---- epilogue (C layout: VGPR r, lanes0-15 -> M=r, lanes16-31 -> M=r+8)
  float gatev = 0.f;
  if constexpr (EPI==1) gatev = gateg[0];
  #pragma unroll
  for (int mt=0; mt<4; ++mt){
    #pragma unroll
    for (int nt=0; nt<4; ++nt){
      #pragma unroll
      for (int r=0; r<8; ++r){
        int row = m0 + wm*64 + mt*16 + r + hk;
        int col = n0 + wn*64 + nt*16 + lm;
        size_t gi = zO + (size_t)row*N + col;
        float a = acc[mt][nt][r];
        if constexpr (EPI==0){
          outh[gi] = (_Float16)a;
        } else if constexpr (EPI==1){
          float xv = xg[gi];
          float hh = (xv - mug[row])*rsg[row]*w1g[col] + b1g[col];
          outp[gi] = xv + hh + gatev*a;
        } else {
          outp[gi] = resg[gi] + a;
        }
      }
    }
  }
}

// ------------------------------------- fused LN3 + low-rank MLP (in-place)
__global__ void channel_mix(float* __restrict__ xio,
                            const float* __restrict__ w3, const float* __restrict__ b3,
                            const float* __restrict__ dw, const float* __restrict__ uw,
                            const float* __restrict__ scale){
  const int tid = threadIdx.x;
  const size_t base = (size_t)blockIdx.x * D_DIM;
  __shared__ float cbuf[D_DIM];
  __shared__ float r1[256], r2[256];
  __shared__ float hid[RANK];
  float v[8]; float s=0.f, q=0.f;
  #pragma unroll
  for (int i=0;i<8;++i){ v[i]=xio[base + tid + i*256]; s+=v[i]; q+=v[i]*v[i]; }
  r1[tid]=s; r2[tid]=q; __syncthreads();
  for (int off=128; off>0; off>>=1){
    if (tid<off){ r1[tid]+=r1[tid+off]; r2[tid]+=r2[tid+off]; }
    __syncthreads();
  }
  const float mean = r1[0]*(1.0f/D_DIM);
  const float var  = r2[0]*(1.0f/D_DIM) - mean*mean;
  const float rs   = rsqrtf(var + 1e-5f);
  #pragma unroll
  for (int i=0;i<8;++i){
    int d = tid + i*256;
    cbuf[d] = (v[i]-mean)*rs*w3[d] + b3[d];
  }
  __syncthreads();
  {  // down proj + gelu: 4 threads per rank
    int r = tid >> 2, qn = tid & 3;
    const float* dwr = dw + (size_t)r*D_DIM + qn*512;
    const float* cb  = cbuf + qn*512;
    float p = 0.f;
    for (int i=0;i<512;++i) p += cb[i]*dwr[i];
    r1[tid] = p; __syncthreads();
    if (qn == 0) hid[r] = gelu_tanh(r1[tid]+r1[tid+1]+r1[tid+2]+r1[tid+3]);
  }
  __syncthreads();
  #pragma unroll
  for (int i=0;i<8;++i){
    int d = tid + i*256;
    const float* uwd = uw + (size_t)d*RANK;
    float a = 0.f;
    #pragma unroll
    for (int r=0;r<RANK;++r) a += hid[r]*uwd[r];
    xio[base+d] = cbuf[d] + scale[d]*a;
  }
}

// --------------------------------------------------------------- launcher
extern "C" void kernel_launch(void* const* d_in, const int* in_sizes, int n_in,
                              void* d_out, int out_size, void* d_ws, size_t ws_size,
                              hipStream_t stream){
  (void)in_sizes; (void)n_in; (void)out_size; (void)ws_size;
  const float* x        = (const float*)d_in[0];
  const float* ln1_w    = (const float*)d_in[1];
  const float* ln1_b    = (const float*)d_in[2];
  const float* ln2_w    = (const float*)d_in[3];
  const float* ln2_b    = (const float*)d_in[4];
  const float* ln3_w    = (const float*)d_in[5];
  const float* ln3_b    = (const float*)d_in[6];
  const float* bind_k   = (const float*)d_in[7];
  const float* unbind_k = (const float*)d_in[8];
  const float* gate     = (const float*)d_in[9];
  const float* fgate    = (const float*)d_in[10];
  const float* down_w   = (const float*)d_in[11];
  const float* up_w     = (const float*)d_in[12];
  const float* scale    = (const float*)d_in[13];
  float* out = (float*)d_out;

  const size_t nBSD = (size_t)B_DIM*S_DIM*D_DIM;
  const size_t nDD  = (size_t)D_DIM*D_DIM;
  float*     t1  = (float*)d_ws;              // x1 (fp32), 128 MiB
  _Float16*  hh  = (_Float16*)(t1 + nBSD);    // h16, later g16    (64 MiB)
  _Float16*  bh  = hh + nBSD;                 // bound16 -> holo16 (64 MiB)
  _Float16*  bt1 = bh + nBSD;                 // circ(bind)^T      (8 MiB)
  _Float16*  bt2 = bt1 + nDD;                 // circ(unbind)^T    (8 MiB)
  _Float16*  kc  = bt2 + nDD;                 // circ(kvec)        (8 MiB)
  float*     mu1 = (float*)(kc + nDD);        // 64 KiB
  float*     rs1 = mu1 + (size_t)B_DIM*S_DIM; // 64 KiB
  float*     kv  = rs1 + (size_t)B_DIM*S_DIM; // 8 KiB
  float*     part= kv + S_DIM;                // 512 KiB

  const int TOK = B_DIM*S_DIM;                // 16384 tokens

  // circulant operand matrices (f16, L2-resident)
  build_circ<<<D_DIM, 256, 0, stream>>>(bind_k,   bt1, +1); // BT[n][k]=bk[(n-k)]
  build_circ<<<D_DIM, 256, 0, stream>>>(unbind_k, bt2, -1); // BT[n][k]=uk[(k-n)]
  kvec_kernel<<<S_DIM/256, 256, 0, stream>>>(fgate, kv);
  build_circ<<<S_DIM, 256, 0, stream>>>(kv, kc, +1);        // A[m][k]=kv[(m-k)]

  // 1) h = LN1(x) -> hh (f16) ; keep per-token mean/rstd for epilogue recompute
  ln_kernel<<<TOK, 256, 0, stream>>>(x, ln1_w, ln1_b, hh, mu1, rs1);
  // 2) bound = circ_conv(h, bind_key) -> bh (f16)
  gemm_wmma<false,0><<<dim3(D_DIM/256, TOK/128, 1), 256, 0, stream>>>(
      hh, bt1, nullptr, bh, TOK, D_DIM, D_DIM, 0, 0,
      nullptr, nullptr, nullptr, nullptr, nullptr, nullptr, nullptr);
  // 3) hologram = cumsum_s(bound), in place on bh
  scan_phase_a<<<512, 256, 0, stream>>>(bh, part);
  scan_phase_c<<<512, 256, 0, stream>>>(bh, part);
  // 4) x1 = x + h + gate*circ_corr(hologram, unbind_key) -> t1 (fused epilogue)
  gemm_wmma<false,1><<<dim3(D_DIM/256, TOK/128, 1), 256, 0, stream>>>(
      bh, bt2, t1, nullptr, TOK, D_DIM, D_DIM, 0, 0,
      x, mu1, rs1, ln1_w, ln1_b, gate, nullptr);
  // 5) g = LN2(x1) -> hh (f16, reuse)
  ln_kernel<<<TOK, 256, 0, stream>>>(t1, ln2_w, ln2_b, hh, nullptr, nullptr);
  // 6) x2 = x1 + Kc @ g  (per batch, dense g transpose-staged) -> d_out
  gemm_wmma<true,2><<<dim3(D_DIM/256, S_DIM/128, B_DIM), 256, 0, stream>>>(
      kc, hh, out, nullptr, S_DIM, D_DIM, S_DIM,
      (size_t)S_DIM*D_DIM, (size_t)S_DIM*D_DIM,
      nullptr, nullptr, nullptr, nullptr, nullptr, nullptr, t1);
  // 7) out = c + scale*up(gelu(down(c))), c = LN3(x2), in place on d_out
  channel_mix<<<TOK, 256, 0, stream>>>(out, ln3_w, ln3_b, down_w, up_w, scale);
}
